// ModelVAE_7198365188165
// MI455X (gfx1250) — matrix operand order repaired and checked by reading, compiled, not run on hardware
//
#include <hip/hip_runtime.h>
#include <math.h>
#include <stdint.h>

// ---------------------------------------------------------------------------
// Types
// ---------------------------------------------------------------------------
typedef __bf16 bf16_t;
typedef __attribute__((ext_vector_type(16))) __bf16 v16bf;
typedef __attribute__((ext_vector_type(8)))  float  v8f;

__device__ __forceinline__ bf16_t f2bf(float f) {
    union { float f; uint32_t u; } v; v.f = f;
    uint32_t r = v.u + 0x7FFFu + ((v.u >> 16) & 1u);   // round-to-nearest-even
    uint16_t h = (uint16_t)(r >> 16);
    bf16_t out; __builtin_memcpy(&out, &h, 2);
    return out;
}
__device__ __forceinline__ float bf2f(bf16_t b) {
    uint16_t h; __builtin_memcpy(&h, &b, 2);
    union { uint32_t u; float f; } v; v.u = ((uint32_t)h) << 16;
    return v.f;
}

// ---------------------------------------------------------------------------
// CDNA5 async global->LDS DMA (ASYNCcnt-tracked), with plain fallback.
// Probe result: param type is `int __attribute__((vector_size(16))) *`.
// ---------------------------------------------------------------------------
#if __has_builtin(__builtin_amdgcn_global_load_async_to_lds_b128) && \
    __has_builtin(__builtin_amdgcn_s_wait_asynccnt)
#define USE_ASYNC_LDS 1
typedef int gcc_v4i __attribute__((__vector_size__(16)));
__device__ __forceinline__ void async_cp16(const bf16_t* g, bf16_t* l) {
    __builtin_amdgcn_global_load_async_to_lds_b128(
        (gcc_v4i*)(void*)g, (gcc_v4i*)(void*)l, 0, 0);
}
#endif

// ---------------------------------------------------------------------------
// fp32 -> bf16, zero-padded to (Rpad x Cpad), same orientation
// ---------------------------------------------------------------------------
__global__ __launch_bounds__(256)
void cvt_pad_kernel(const float* __restrict__ in, bf16_t* __restrict__ out,
                    int R, int C, int Rpad, int Cpad) {
    long total = (long)Rpad * (long)Cpad;
    for (long i = (long)blockIdx.x * blockDim.x + threadIdx.x; i < total;
         i += (long)gridDim.x * blockDim.x) {
        long r = i / Cpad, c = i % Cpad;
        float v = (r < R && c < C) ? in[r * (long)C + c] : 0.0f;
        out[i] = f2bf(v);
    }
}

// ---------------------------------------------------------------------------
// fp32 W[K x N] -> bf16 Wt[Npad x Kpad] (transposed, zero-padded).
// ---------------------------------------------------------------------------
__global__ __launch_bounds__(256)
void cvt_padT_kernel(const float* __restrict__ in, bf16_t* __restrict__ out,
                     int K, int N, int Kpad, int Npad) {
    long total = (long)Npad * (long)Kpad;
    for (long i = (long)blockIdx.x * blockDim.x + threadIdx.x; i < total;
         i += (long)gridDim.x * blockDim.x) {
        long n = i / Kpad, k = i % Kpad;
        float v = (n < N && k < K) ? in[k * (long)N + n] : 0.0f;
        out[i] = f2bf(v);
    }
}

// ---------------------------------------------------------------------------
// Tiled bf16 WMMA GEMM:  out = act(A[MxK] @ W[KxN] + bias)
//   A : [M x K] row-major bf16 (K multiple of 32, M multiple of 128)
//   Wt: [Npad x K] bf16 = W transposed, Npad = gridDim.x*128 (zero padded)
// Block = 256 threads (8 waves).  Block tile 128x128, K-panel 32.
// Wave tile 32x64 = 2x4 v_wmma_f32_16x16x32_bf16.  Double-buffered LDS:
//  - async path: GLOBAL_LOAD_ASYNC_TO_LDS_B128, drained by s_wait_asynccnt 0
//    immediately before the panel barrier (latency covered by WMMA burst).
//  - fallback:   register-pipelined (loads issued post-barrier for panel k+1,
//    ds_store + loadcnt wait deferred to the top of the next iteration).
// ---------------------------------------------------------------------------
#define BM 128
#define BN 128
#define BK 32
#define LDT 40   // LDS row stride in bf16 (80 B: 16B-aligned + bank-conflict pad)

__global__ __launch_bounds__(256)
void gemm_bf16_wmma(const bf16_t* __restrict__ A, const bf16_t* __restrict__ Wt,
                    const float* __restrict__ bias,
                    float* __restrict__ outF, bf16_t* __restrict__ outB,
                    int M, int N, int K, int relu) {
    __shared__ __align__(16) bf16_t As[2][BM * LDT];
    __shared__ __align__(16) bf16_t Bs[2][BN * LDT];

    const int tid  = threadIdx.x;
    const int lane = tid & 31;
    const int wave = tid >> 5;
    const int wr   = wave & 3;     // row quadrant (0..3)  -> 32 rows
    const int wc   = wave >> 2;    // col half     (0..1)  -> 64 cols
    const int rowBase = blockIdx.y * BM;
    const int colBase = blockIdx.x * BN;
    const int lmod = lane & 15;
    const int ldiv = lane >> 4;

    // Staging map: thread loads 16 contiguous halves of one row (2 x b128)
    const int srow = tid >> 1;          // 0..127
    const int scp  = (tid & 1) << 4;    // 0 / 16
    const bf16_t* agBase = A  + (long)(rowBase + srow) * K + scp;
    const bf16_t* bgBase = Wt + (long)(colBase + srow) * K + scp;

    v8f acc[2][4];
    #pragma unroll
    for (int sm = 0; sm < 2; ++sm)
        #pragma unroll
        for (int sn = 0; sn < 4; ++sn) acc[sm][sn] = (v8f){};

    union Frag { v16bf v; uint4 q[2]; };
    int cur = 0;

#ifdef USE_ASYNC_LDS
    auto stage = [&](int buf, int k0) {
        async_cp16(agBase + k0,     &As[buf][srow * LDT + scp]);
        async_cp16(agBase + k0 + 8, &As[buf][srow * LDT + scp + 8]);
        async_cp16(bgBase + k0,     &Bs[buf][srow * LDT + scp]);
        async_cp16(bgBase + k0 + 8, &Bs[buf][srow * LDT + scp + 8]);
    };
    stage(0, 0);
#else
    uint4 ra0, ra1, rb0, rb1;
    auto gload = [&](int k0) {
        ra0 = ((const uint4*)(agBase + k0))[0];
        ra1 = ((const uint4*)(agBase + k0))[1];
        rb0 = ((const uint4*)(bgBase + k0))[0];
        rb1 = ((const uint4*)(bgBase + k0))[1];
    };
    auto lstore = [&](int buf) {
        *(uint4*)&As[buf][srow * LDT + scp]     = ra0;
        *(uint4*)&As[buf][srow * LDT + scp + 8] = ra1;
        *(uint4*)&Bs[buf][srow * LDT + scp]     = rb0;
        *(uint4*)&Bs[buf][srow * LDT + scp + 8] = rb1;
    };
    gload(0);
#endif

    for (int k0 = 0; k0 < K; k0 += BK) {
#ifdef USE_ASYNC_LDS
        __builtin_amdgcn_s_wait_asynccnt(0);   // panel `cur` landed in LDS
        __syncthreads();                       // publish across waves
        if (k0 + BK < K) {
            stage(cur ^ 1, k0 + BK);           // fire-and-forget DMA
            if (k0 + 2 * BK < K)
                __builtin_prefetch(agBase + k0 + 2 * BK, 0, 1);
        }
#else
        lstore(cur);                           // loadcnt wait lands here, after
        __syncthreads();                       //  a full panel of WMMAs
        if (k0 + BK < K) {
            gload(k0 + BK);                    // issue early, consume later
            if (k0 + 2 * BK < K)
                __builtin_prefetch(agBase + k0 + 2 * BK, 0, 1);
        }
#endif

        // fragments (ISA 16-bit A 16x32 / B 32x16 lane layouts)
        Frag af[2], bfg[4];
        #pragma unroll
        for (int sm = 0; sm < 2; ++sm) {
            const int r = wr * 32 + sm * 16 + lmod;
            af[sm].q[0] = *(const uint4*)&As[cur][r * LDT + (ldiv << 3)];
            af[sm].q[1] = *(const uint4*)&As[cur][r * LDT + 16 + (ldiv << 3)];
        }
        #pragma unroll
        for (int sn = 0; sn < 4; ++sn) {
            const int n = wc * 64 + sn * 16 + lmod;
            bfg[sn].q[0] = *(const uint4*)&Bs[cur][n * LDT + (ldiv << 3)];
            bfg[sn].q[1] = *(const uint4*)&Bs[cur][n * LDT + 16 + (ldiv << 3)];
        }
        #pragma unroll
        for (int sm = 0; sm < 2; ++sm)
            #pragma unroll
            for (int sn = 0; sn < 4; ++sn)
                acc[sm][sn] = __builtin_amdgcn_wmma_f32_16x16x32_bf16(
                    false, af[sm].v, false, bfg[sn].v,
                    (short)0, acc[sm][sn], false, false);
        cur ^= 1;
    }

    // ---- epilogue: bias + optional relu, dual fp32 / bf16 stores ----
    #pragma unroll
    for (int sm = 0; sm < 2; ++sm) {
        #pragma unroll
        for (int sn = 0; sn < 4; ++sn) {
            const int mrow = rowBase + wr * 32 + sm * 16 + (ldiv << 3);
            const int ncol = colBase + wc * 64 + sn * 16 + lmod;
            if (ncol < N) {
                const float bv = bias ? bias[ncol] : 0.0f;
                #pragma unroll
                for (int i = 0; i < 8; ++i) {
                    float v = acc[sm][sn][i] + bv;
                    if (relu) v = fmaxf(v, 0.0f);
                    const long off = (long)(mrow + i) * N + ncol;
                    if (outF) outF[off] = v;
                    if (outB) outB[off] = f2bf(v);
                }
            }
        }
    }
}

// ---------------------------------------------------------------------------
// kappa = softplus(h @ conc_w + conc_b) + 1   (N==1 GEMV, one wave per row)
// ---------------------------------------------------------------------------
__global__ __launch_bounds__(256)
void kappa_kernel(const bf16_t* __restrict__ h, const float* __restrict__ w,
                  const float* __restrict__ b, float* __restrict__ kout, int Hd) {
    const int row  = blockIdx.x * 8 + (threadIdx.x >> 5);
    const int lane = threadIdx.x & 31;
    const bf16_t* hr = h + (long)row * Hd;
    float s = 0.0f;
    for (int j = lane; j < Hd; j += 32) s += bf2f(hr[j]) * w[j];
    #pragma unroll
    for (int m = 16; m; m >>= 1) s += __shfl_xor(s, m, 32);
    if (lane == 0) {
        float v  = s + b[0];
        float sp = (v > 20.0f) ? v : log1pf(expf(v));
        kout[row] = sp + 1.0f;
    }
}

// ---------------------------------------------------------------------------
// von Mises rejection sampler (Best-Fisher, fixed 32 rounds).
// Counter-based hash RNG (JAX threefry not bit-reproducible; no runtime check).
// ---------------------------------------------------------------------------
__device__ __forceinline__ float rnd01(uint32_t x, uint32_t y, uint32_t z) {
    uint32_t s = x * 0x9E3779B1u ^ y * 0x85EBCA77u ^ z * 0xC2B2AE3Du;
    s ^= s >> 16; s *= 0x7FEB352Du;
    s ^= s >> 15; s *= 0x846CA68Bu;
    s ^= s >> 16;
    return (float)(s >> 8) * (1.0f / 16777216.0f) + (0.5f / 16777216.0f);
}

__global__ __launch_bounds__(256)
void vm_sampler(const float* __restrict__ mu, const float* __restrict__ kappa,
                float* __restrict__ theta_out, int total) {
    const float PI_F = 3.14159265358979323846f;
    int idx = blockIdx.x * 256 + threadIdx.x;
    if (idx >= total) return;
    const float loc = mu[idx];
    const float kap = kappa[idx >> 6];

    const float tau = 1.0f + sqrtf(1.0f + 4.0f * kap * kap);
    const float rho = (tau - sqrtf(2.0f * tau)) / (2.0f * kap);
    const float r   = (1.0f + rho * rho) / (2.0f * rho);

    float theta = 0.0f;
    bool  done  = false;
    #pragma unroll 4
    for (int it = 0; it < 32; ++it) {
        const float u1 = rnd01((uint32_t)idx, (uint32_t)it, 0u);
        const float u2 = rnd01((uint32_t)idx, (uint32_t)it, 1u);
        const float u3 = rnd01((uint32_t)idx, (uint32_t)it, 2u);
        const float zc = cosf(PI_F * u1);
        const float f  = (1.0f + r * zc) / (r + zc);
        const float c  = kap * (r - f);
        const bool acc = (c * (2.0f - c) - u2 > 0.0f) ||
                         (logf(c / u2) + 1.0f - c >= 0.0f);
        const float fcl  = fminf(fmaxf(f, -1.0f), 1.0f);
        const float prop = copysignf(acosf(fcl), u3 - 0.5f);
        if (!done && acc) theta = prop;
        done = done || acc;
    }
    float t = fmodf(theta + loc + PI_F, 2.0f * PI_F);
    if (t < 0.0f) t += 2.0f * PI_F;
    theta_out[idx] = t - PI_F;
}

// ---------------------------------------------------------------------------
// z_c = ifft(exp(i*theta_s), ortho) with conjugate-symmetric phases:
//   z[n] = (1 + (-1)^n + 2*sum_{k=1}^{63} cos(theta_k + pi*n*k/64)) / sqrt(128)
//   imag(z) == 0.  Output interleaved (re, im) pairs as bf16 [B x 256].
// ---------------------------------------------------------------------------
__global__ __launch_bounds__(256)
void ifft_phase_kernel(const float* __restrict__ theta, bf16_t* __restrict__ zr) {
    __shared__ float th[2][64];
    const int sub = threadIdx.x >> 7;     // row within block
    const int n   = threadIdx.x & 127;    // output index 0..127
    const long row = (long)blockIdx.x * 2 + sub;
    if (n < 64) th[sub][n] = theta[row * 64 + n];
    __syncthreads();

    const float step = 3.14159265358979323846f / 64.0f;
    float sum = 1.0f + ((n & 1) ? -1.0f : 1.0f);
    #pragma unroll 7
    for (int k = 1; k < 64; ++k)
        sum += 2.0f * cosf(th[sub][k] + step * (float)(n * k));

    const float scale = 0.0883883476483184f;   // 1/sqrt(128)
    zr[row * 256 + 2 * n]     = f2bf(sum * scale);
    zr[row * 256 + 2 * n + 1] = f2bf(0.0f);
}

// ---------------------------------------------------------------------------
// Host launcher
// ---------------------------------------------------------------------------
extern "C" void kernel_launch(void* const* d_in, const int* in_sizes, int n_in,
                              void* d_out, int out_size, void* d_ws, size_t ws_size,
                              hipStream_t stream) {
    (void)in_sizes; (void)n_in; (void)out_size;

    const int Bn = 16384, Hd = 1024, Zn = 64;
    const int K1 = 784, K1p = 800;      // layer-1 K padded to multiple of 32

    // inputs (setup_inputs dict order)
    const float* x      = (const float*)d_in[0];
    const float* enc_w1 = (const float*)d_in[1];
    const float* enc_b1 = (const float*)d_in[2];
    const float* enc_w2 = (const float*)d_in[3];
    const float* enc_b2 = (const float*)d_in[4];
    const float* mu_w   = (const float*)d_in[5];
    const float* mu_b   = (const float*)d_in[6];
    const float* conc_w = (const float*)d_in[7];
    const float* conc_b = (const float*)d_in[8];
    const float* pre_w  = (const float*)d_in[9];
    const float* pre_b  = (const float*)d_in[10];
    const float* dec_w1 = (const float*)d_in[11];
    const float* dec_b1 = (const float*)d_in[12];
    const float* dec_w2 = (const float*)d_in[13];
    const float* dec_b2 = (const float*)d_in[14];
    const float* dec_w3 = (const float*)d_in[15];
    const float* dec_b3 = (const float*)d_in[16];

    // outputs: mu [B,Z] | kappa [B,1] | z [B,Z] | x_recon [B,784]
    float* mu_o  = (float*)d_out;
    float* kap_o = mu_o + (long)Bn * Zn;
    float* z_o   = kap_o + Bn;
    float* xr_o  = z_o + (long)Bn * Zn;

    // workspace partition (lifetime-aliased)
    char* p = (char*)d_ws;
    auto alloc = [&](size_t bytes) -> char* {
        char* r = p; p += (bytes + 255) & ~(size_t)255; return r;
    };
    // transposed weights  Wt[Npad x Kpad]
    bf16_t* w1t   = (bf16_t*)alloc((size_t)2048 * K1p * 2);
    bf16_t* w2t   = (bf16_t*)alloc((size_t)1024 * 2048 * 2);
    bf16_t* muwt  = (bf16_t*)alloc((size_t)128 * 1024 * 2);
    bf16_t* prewt = (bf16_t*)alloc((size_t)128 * 256 * 2);
    bf16_t* dw1t  = (bf16_t*)alloc((size_t)1024 * 64 * 2);
    bf16_t* dw2t  = (bf16_t*)alloc((size_t)2048 * 1024 * 2);
    bf16_t* dw3t  = (bf16_t*)alloc((size_t)896 * 2048 * 2);
    char* regA    = alloc((size_t)Bn * K1p * 2);       // xb; later theta/zr/zb
    char* regB    = alloc((size_t)Bn * 2048 * 2);      // h1b; later d2b
    char* regC    = alloc((size_t)Bn * 1024 * 2);      // hb;  later d1b
    if ((size_t)(p - (char*)d_ws) > ws_size) return;   // insufficient scratch

    bf16_t* xb    = (bf16_t*)regA;
    float*  theta = (float*)regA;                       // xb dead after enc1
    bf16_t* zrb   = (bf16_t*)(regA + (size_t)Bn * Zn * 4);
    bf16_t* zb    = (bf16_t*)(regA + (size_t)Bn * Zn * 4 + (size_t)Bn * 256 * 2);
    bf16_t* h1b   = (bf16_t*)regB;
    bf16_t* d2b   = (bf16_t*)regB;                      // h1b dead after enc2
    bf16_t* hb    = (bf16_t*)regC;
    bf16_t* d1b   = (bf16_t*)regC;                      // hb dead after kappa

    // ---- conversions: activations padded row-major, weights transposed ----
    cvt_pad_kernel <<<2048, 256, 0, stream>>>(x,      xb,    Bn, K1, Bn, K1p);
    cvt_padT_kernel<<<2048, 256, 0, stream>>>(enc_w1, w1t,   K1, 2048, K1p, 2048);
    cvt_padT_kernel<<<2048, 256, 0, stream>>>(enc_w2, w2t,   2048, 1024, 2048, 1024);
    cvt_padT_kernel<<<256,  256, 0, stream>>>(mu_w,   muwt,  1024, 64, 1024, 128);
    cvt_padT_kernel<<<64,   256, 0, stream>>>(pre_w,  prewt, 256, 64, 256, 128);
    cvt_padT_kernel<<<256,  256, 0, stream>>>(dec_w1, dw1t,  64, 1024, 64, 1024);
    cvt_padT_kernel<<<2048, 256, 0, stream>>>(dec_w2, dw2t,  1024, 2048, 1024, 2048);
    cvt_padT_kernel<<<2048, 256, 0, stream>>>(dec_w3, dw3t,  2048, 784, 2048, 896);

    const dim3 blk(256);
    // ---- encoder ----
    gemm_bf16_wmma<<<dim3(2048 / BN, Bn / BM), blk, 0, stream>>>(
        xb, w1t, enc_b1, nullptr, h1b, Bn, 2048, K1p, 1);
    gemm_bf16_wmma<<<dim3(1024 / BN, Bn / BM), blk, 0, stream>>>(
        h1b, w2t, enc_b2, nullptr, hb, Bn, 1024, 2048, 1);
    gemm_bf16_wmma<<<dim3(1, Bn / BM), blk, 0, stream>>>(
        hb, muwt, mu_b, mu_o, nullptr, Bn, 64, 1024, 0);
    kappa_kernel<<<Bn / 8, 256, 0, stream>>>(hb, conc_w, conc_b, kap_o, Hd);

    // ---- reparameterize: von Mises sample -> phase iFFT ----
    vm_sampler<<<(Bn * Zn) / 256, 256, 0, stream>>>(mu_o, kap_o, theta, Bn * Zn);
    ifft_phase_kernel<<<Bn / 2, 256, 0, stream>>>(theta, zrb);
    gemm_bf16_wmma<<<dim3(1, Bn / BM), blk, 0, stream>>>(
        zrb, prewt, pre_b, z_o, zb, Bn, 64, 256, 0);

    // ---- decoder ----
    gemm_bf16_wmma<<<dim3(1024 / BN, Bn / BM), blk, 0, stream>>>(
        zb, dw1t, dec_b1, nullptr, d1b, Bn, 1024, 64, 1);
    gemm_bf16_wmma<<<dim3(2048 / BN, Bn / BM), blk, 0, stream>>>(
        d1b, dw2t, dec_b2, nullptr, d2b, Bn, 2048, 1024, 1);
    gemm_bf16_wmma<<<dim3(896 / BN, Bn / BM), blk, 0, stream>>>(
        d2b, dw3t, dec_b3, xr_o, nullptr, Bn, 784, 2048, 0);
}
